// DBRXMoE_23587960390189
// MI455X (gfx1250) — matrix operand order repaired and checked by reading, not compiled, not run on hardware
//
#include <hip/hip_runtime.h>
#include <math.h>

// ---------------- problem constants (from reference) ----------------
constexpr int NTOK = 1024;   // B*T
constexpr int C    = 2048;   // d_model
constexpr int H    = 3584;   // ffn hidden
constexpr int E    = 16;     // experts
constexpr int K    = 4;      // top-k

typedef __attribute__((ext_vector_type(16))) __bf16 v16bf;
typedef __attribute__((ext_vector_type(8)))  __bf16 v8bf;
typedef __attribute__((ext_vector_type(8)))  float  v8f;

// ---------------- fragment loaders ----------------
// B fragment (32x16, bf16): lane half 0 -> K 0..15, half 1 -> K 16..31,
// column n = lane&15. Caller passes p = rowBase + c0 + 16*(lane>>4).
__device__ __forceinline__ v16bf frag_b_f32(const float* __restrict__ p) {
  float4 q0 = *(const float4*)(p + 0);
  float4 q1 = *(const float4*)(p + 4);
  float4 q2 = *(const float4*)(p + 8);
  float4 q3 = *(const float4*)(p + 12);
  v16bf f;
  f[0]=(__bf16)q0.x;  f[1]=(__bf16)q0.y;  f[2]=(__bf16)q0.z;  f[3]=(__bf16)q0.w;
  f[4]=(__bf16)q1.x;  f[5]=(__bf16)q1.y;  f[6]=(__bf16)q1.z;  f[7]=(__bf16)q1.w;
  f[8]=(__bf16)q2.x;  f[9]=(__bf16)q2.y;  f[10]=(__bf16)q2.z; f[11]=(__bf16)q2.w;
  f[12]=(__bf16)q3.x; f[13]=(__bf16)q3.y; f[14]=(__bf16)q3.z; f[15]=(__bf16)q3.w;
  return f;
}

// A fragment (16x32, bf16): lane half 0 -> K 0..15, half 1 -> K 16..31,
// row m = lane&15. Caller passes p = rowBase + c0 + 16*(lane>>4).
__device__ __forceinline__ v16bf frag_a_bf16(const __bf16* __restrict__ p) {
  v8bf lo = *(const v8bf*)(p);
  v8bf hi = *(const v8bf*)(p + 8);
  v16bf f;
#pragma unroll
  for (int i = 0; i < 8; ++i) { f[i] = lo[i]; f[8 + i] = hi[i]; }
  return f;
}

// silu via fast hardware reciprocal (v_rcp_f32), avoids IEEE div expansion
__device__ __forceinline__ float silu_f(float g) {
  return g * __builtin_amdgcn_rcpf(1.0f + __expf(-g));
}

// ---------------- kernel 0: zero counters + x -> bf16 ----------------
__global__ void moe_init_kernel(const float* __restrict__ x,
                                __bf16* __restrict__ xbf,
                                int* __restrict__ counters) {
  int i = blockIdx.x * blockDim.x + threadIdx.x;
  if (i < E) counters[i] = 0;
  for (int j = i; j < NTOK * C; j += gridDim.x * blockDim.x)
    xbf[j] = (__bf16)x[j];
}

// ---------------- kernel 1: routing (one wave32 per token) ----------------
__global__ void moe_route_kernel(const float* __restrict__ x,
                                 const float* __restrict__ gate,
                                 int* __restrict__ counters,
                                 int* __restrict__ lists,
                                 float* __restrict__ wslot) {
  const int wave = threadIdx.x >> 5;
  const int lane = threadIdx.x & 31;
  const int n = blockIdx.x * 8 + wave;
  if (n >= NTOK) return;

  float acc[E];
#pragma unroll
  for (int e = 0; e < E; ++e) acc[e] = 0.0f;

  const float* xr = x + (size_t)n * C;
  for (int c = lane; c < C; c += 32) {
    float xv = xr[c];
#pragma unroll
    for (int e = 0; e < E; ++e) acc[e] += xv * gate[e * C + c];
  }
  // wave32 allreduce
#pragma unroll
  for (int e = 0; e < E; ++e) {
#pragma unroll
    for (int off = 16; off > 0; off >>= 1)
      acc[e] += __shfl_xor(acc[e], off, 32);
  }

  if (lane == 0) {
    int   ti[K];
    float tv[K];
    unsigned used = 0;
#pragma unroll
    for (int k = 0; k < K; ++k) {
      int bi = 0; float bv = -INFINITY;
#pragma unroll
      for (int e = 0; e < E; ++e) {
        if (!((used >> e) & 1u) && acc[e] > bv) { bv = acc[e]; bi = e; }
      }
      used |= (1u << bi);
      ti[k] = bi; tv[k] = bv;
    }
    float mx = tv[0];
    float w[K]; float sum = 0.0f;
#pragma unroll
    for (int k = 0; k < K; ++k) { w[k] = __expf(tv[k] - mx); sum += w[k]; }
    float inv = 1.0f / sum;
#pragma unroll
    for (int k = 0; k < K; ++k) {
      int slot = n * K + k;
      wslot[slot] = w[k] * inv;
      int pos = atomicAdd(&counters[ti[k]], 1);
      lists[ti[k] * NTOK + pos] = slot;
    }
  }
}

// ---------------- kernel 2: grouped up/gate GEMM + SwiGLU (WMMA bf16) ----------------
// Block: 4 waves, each wave owns 64 M-rows (4 A-frags) x 16 h-columns x {up,gate}.
// One B-fragment pair per K-step feeds 8 WMMAs.
__global__ void __launch_bounds__(128)
moe_upgate_kernel(const float* __restrict__ Wup,
                  const float* __restrict__ Wgt,
                  const __bf16* __restrict__ xbf,
                  const int* __restrict__ counters,
                  const int* __restrict__ lists,
                  const float* __restrict__ wslot,
                  __bf16* __restrict__ hbuf) {
  const int e   = blockIdx.y;
  const int cnt = counters[e];
  if (cnt == 0) return;

  const int h0   = blockIdx.x * 16;
  const int wave = threadIdx.x >> 5;
  const int lane = threadIdx.x & 31;
  const int nL   = lane & 15;
  const int half = lane >> 4;
  const int kbA  = half * 16;
  const int kbB  = half * 16;

  const float* upRow = Wup + ((size_t)e * H + h0 + nL) * C;
  const float* gtRow = Wgt + ((size_t)e * H + h0 + nL) * C;
  const int*   lst   = lists + e * NTOK;

  for (int m0 = wave * 64; m0 < cnt; m0 += 256) {
    const __bf16* xp[4];
#pragma unroll
    for (int f = 0; f < 4; ++f) {
      int r = m0 + f * 16 + nL;
      if (r > cnt - 1) r = cnt - 1;
      xp[f] = xbf + (size_t)(lst[r] >> 2) * C;
    }

    v8f accU[4] = {{}, {}, {}, {}};
    v8f accG[4] = {{}, {}, {}, {}};
    for (int c0 = 0; c0 < C; c0 += 32) {
      __builtin_prefetch(upRow + c0 + 256, 0, 1);
      __builtin_prefetch(gtRow + c0 + 256, 0, 1);
      v16bf fbU = frag_b_f32(upRow + c0 + kbB);
      v16bf fbG = frag_b_f32(gtRow + c0 + kbB);
#pragma unroll
      for (int f = 0; f < 4; ++f) {
        v16bf fa = frag_a_bf16(xp[f] + c0 + kbA);
        accU[f] = __builtin_amdgcn_wmma_f32_16x16x32_bf16(false, fa, false, fbU,
                                                          (short)0, accU[f], false, false);
        accG[f] = __builtin_amdgcn_wmma_f32_16x16x32_bf16(false, fa, false, fbG,
                                                          (short)0, accG[f], false, false);
      }
    }

    const int h = h0 + nL;
#pragma unroll
    for (int f = 0; f < 4; ++f) {
#pragma unroll
      for (int r = 0; r < 8; ++r) {
        int m = m0 + f * 16 + r + 8 * half;
        if (m < cnt) {
          int slot = lst[m];
          float hv = accU[f][r] * silu_f(accG[f][r]) * wslot[slot];
          hbuf[(size_t)slot * H + h] = (__bf16)hv;
        }
      }
    }
  }
}

// ---------------- kernel 3: grouped down GEMM (WMMA bf16) ----------------
__global__ void __launch_bounds__(128)
moe_down_kernel(const float* __restrict__ Wdn,
                const __bf16* __restrict__ hbuf,
                const int* __restrict__ counters,
                const int* __restrict__ lists,
                float* __restrict__ part) {
  const int e   = blockIdx.y;
  const int cnt = counters[e];
  if (cnt == 0) return;

  const int c0t  = blockIdx.x * 16;
  const int wave = threadIdx.x >> 5;
  const int lane = threadIdx.x & 31;
  const int nL   = lane & 15;
  const int half = lane >> 4;
  const int kbA  = half * 16;
  const int kbB  = half * 16;

  const float* dRow = Wdn + ((size_t)e * C + c0t + nL) * H;
  const int*   lst  = lists + e * NTOK;

  for (int m0 = wave * 64; m0 < cnt; m0 += 256) {
    const __bf16* hp[4];
#pragma unroll
    for (int f = 0; f < 4; ++f) {
      int r = m0 + f * 16 + nL;
      if (r > cnt - 1) r = cnt - 1;
      hp[f] = hbuf + (size_t)lst[r] * H;
    }

    v8f acc[4] = {{}, {}, {}, {}};
    for (int hh = 0; hh < H; hh += 32) {
      __builtin_prefetch(dRow + hh + 256, 0, 1);
      v16bf fb = frag_b_f32(dRow + hh + kbB);
#pragma unroll
      for (int f = 0; f < 4; ++f) {
        v16bf fa = frag_a_bf16(hp[f] + hh + kbA);
        acc[f] = __builtin_amdgcn_wmma_f32_16x16x32_bf16(false, fa, false, fb,
                                                         (short)0, acc[f], false, false);
      }
    }

    const int c = c0t + nL;
#pragma unroll
    for (int f = 0; f < 4; ++f) {
#pragma unroll
      for (int r = 0; r < 8; ++r) {
        int m = m0 + f * 16 + r + 8 * half;
        if (m < cnt) part[(size_t)lst[m] * C + c] = acc[f][r];
      }
    }
  }
}

// ---------------- kernel 4: combine 4 slots per token ----------------
__global__ void moe_combine_kernel(const float* __restrict__ part,
                                   float* __restrict__ y) {
  int i = blockIdx.x * blockDim.x + threadIdx.x;
  if (i >= NTOK * C) return;
  int n = i >> 11;          // / C (C = 2048)
  int c = i & (C - 1);
  const float* p = part + (size_t)n * K * C + c;
  y[i] = p[0] + p[C] + p[2 * C] + p[3 * C];
}

// ---------------- host launch ----------------
extern "C" void kernel_launch(void* const* d_in, const int* in_sizes, int n_in,
                              void* d_out, int out_size, void* d_ws, size_t ws_size,
                              hipStream_t stream) {
  (void)in_sizes; (void)n_in; (void)out_size; (void)ws_size;

  const float* x     = (const float*)d_in[0];   // [N, C]
  const float* Wup   = (const float*)d_in[1];   // [E, H, C]
  const float* Wgt   = (const float*)d_in[2];   // [E, H, C]
  const float* Wdn   = (const float*)d_in[3];   // [E, C, H]
  const float* gate  = (const float*)d_in[4];   // [E, C]
  float*       y     = (float*)d_out;           // [N, C]

  char* ws = (char*)d_ws;
  size_t off = 0;
  auto wsAlloc = [&](size_t bytes) -> void* {
    void* p = ws + off;
    off += (bytes + 255) & ~(size_t)255;
    return p;
  };
  __bf16* xbf      = (__bf16*)wsAlloc((size_t)NTOK * C * 2);        //  4 MB
  __bf16* hbuf     = (__bf16*)wsAlloc((size_t)NTOK * K * H * 2);    // 28 MB
  float*  part     = (float*) wsAlloc((size_t)NTOK * K * C * 4);    // 32 MB
  int*    counters = (int*)   wsAlloc((size_t)E * 4);
  int*    lists    = (int*)   wsAlloc((size_t)E * NTOK * 4);
  float*  wslot    = (float*) wsAlloc((size_t)NTOK * K * 4);

  moe_init_kernel<<<(NTOK * C) / 256, 256, 0, stream>>>(x, xbf, counters);
  moe_route_kernel<<<NTOK / 8, 256, 0, stream>>>(x, gate, counters, lists, wslot);

  dim3 g2(H / 16, E);   // 224 x 16
  moe_upgate_kernel<<<g2, 128, 0, stream>>>(Wup, Wgt, xbf, counters, lists, wslot, hbuf);

  dim3 g3(C / 16, E);   // 128 x 16
  moe_down_kernel<<<g3, 128, 0, stream>>>(Wdn, hbuf, counters, lists, part);

  moe_combine_kernel<<<(NTOK * C + 255) / 256, 256, 0, stream>>>(part, y);
}
